// sLSTMLayer_22952305230490
// MI455X (gfx1250) — matrix operand order repaired and checked by reading, not compiled, not run on hardware
//
#include <hip/hip_runtime.h>
#include <math.h>
#include <stdint.h>

typedef __attribute__((ext_vector_type(16))) __bf16 v16bf;
typedef __attribute__((ext_vector_type(8)))  __bf16 v8bf;
typedef __attribute__((ext_vector_type(8)))  float  v8f;
typedef __attribute__((address_space(3)))    __bf16 lds_bf16;

#define Bz   8
#define Sz   4096
#define Dz   768
#define NHz  4
#define DHz  192
#define KCz  4
#define NCOL 768            // 4 gates * DHz
#define EPSF 1e-5f

// ---- A-operand fragment loader -------------------------------------------
// 16-bit A-matrix 16x32 fragment, contiguous K order (matches B): lane holds
// row M=lane%16 with K = 16*kh..16*kh+15 (kh = lane>>4).  Each lane loads two
// contiguous 16-byte chunks at K-offsets 16*kh and 16*kh+8 within the K-slab.
__device__ __forceinline__ v16bf load_a_frag(const __bf16* rowbase, int kb, int kh) {
  const __bf16* p0 = rowbase + kb * 32 + kh * 16;
  v8bf lo = *(const v8bf*)p0;
  v8bf hi = *(const v8bf*)(p0 + 8);
  return __builtin_shufflevector(lo, hi, 0,1,2,3,4,5,6,7,8,9,10,11,12,13,14,15);
}

// ---- Kernel 1: depthwise causal conv + SiLU ------------------------------
__global__ void k_conv_silu(const float* __restrict__ x, const float* __restrict__ cw,
                            const float* __restrict__ cb, float* __restrict__ xc) {
  long idx = (long)blockIdx.x * blockDim.x + threadIdx.x;
  const long total = (long)Bz * Sz * Dz;
  if (idx >= total) return;
  int d = (int)(idx % Dz);
  long bs = idx / Dz;
  int s = (int)(bs % Sz);
  float acc = cb[d];
#pragma unroll
  for (int k = 0; k < KCz; ++k) {
    int sp = s - (KCz - 1) + k;
    if (sp >= 0) acc += x[idx + (long)(sp - s) * Dz] * cw[d * KCz + k];
  }
  xc[idx] = acc / (1.0f + __expf(-acc));   // silu
}

// ---- Kernel 2: pack weights to bf16 in WMMA-B-friendly (K-contiguous) form
// wbf[g][h][o][d] : B operand for projections (o = N column, d = K contiguous)
// rbf[h][n][d]    : B operand for recurrence, n = g*DHz + e, d = K contiguous
__global__ void k_pack(const float* __restrict__ wf, const float* __restrict__ wi,
                       const float* __restrict__ wz, const float* __restrict__ wo,
                       const float* __restrict__ rk,
                       __bf16* __restrict__ wbf, __bf16* __restrict__ rbf) {
  int tid = blockIdx.x * 256 + threadIdx.x;
  const int WN = NHz * DHz * DHz;            // 147456 per gate
  if (tid < 4 * WN) {
    int g = tid / WN, r = tid % WN;
    const float* src = (g == 0) ? wf : (g == 1) ? wi : (g == 2) ? wz : wo;
    wbf[tid] = (__bf16)src[r];
  }
  const int RN = NHz * NCOL * DHz;           // 589824
  if (tid < RN) {
    int h   = tid / (NCOL * DHz);
    int rem = tid % (NCOL * DHz);
    int n = rem / DHz, d = rem % DHz;
    int g = n / DHz,  e = n % DHz;
    // r_kernel shape (NH, DH, 4, DH): [h][d][g][e]
    rbf[tid] = (__bf16)rk[((h * DHz + d) * 4 + g) * DHz + e];
  }
}

// ---- Kernel 3: headwise projections via WMMA -----------------------------
// Computes Wx[s][h][g][b][e] = sum_d A[token][h*DHz+d] * w_g_h[e][d]
// g=0:(x_conv,w_f) g=1:(x_conv,w_i) g=2:(x,w_z) g=3:(x,w_o)
__global__ __launch_bounds__(256) void k_proj(const float* __restrict__ x,
                                              const float* __restrict__ xc,
                                              const __bf16* __restrict__ wbf,
                                              float* __restrict__ Wx) {
  __shared__ __bf16 Axc[16 * Dz];        // 24 KB
  __shared__ __bf16 Ax [16 * Dz];        // 24 KB
  __shared__ float  cstage[8 * 256];     // 8 KB: per-wave 16x16 C staging
  const int tid = threadIdx.x;
  const int m0  = blockIdx.x * 16;            // 16 tokens per block
  for (int i = tid; i < 16 * Dz; i += 256) {
    int m = i / Dz, c = i % Dz;
    long off = (long)(m0 + m) * Dz + c;
    Ax [i] = (__bf16)x [off];
    Axc[i] = (__bf16)xc[off];
  }
  __syncthreads();
  const int lane = tid & 31, w = tid >> 5;
  const int c4 = lane & 15, kh = lane >> 4;
  float* stg = &cstage[w * 256];
  // 192 output tiles (4g x 4h x 12n) -> 24 per wave
  for (int ti = w * 24; ti < w * 24 + 24; ++ti) {
    int g = ti / 48;
    int h = (ti % 48) / 12;
    int n = ti % 12;
    const __bf16* A    = (g < 2) ? Axc : Ax;
    const __bf16* arow = A + c4 * Dz + h * DHz;
    const __bf16* brow = wbf + (size_t)((g * NHz + h) * DHz + n * 16 + c4) * DHz;
    v8f acc = {0.f,0.f,0.f,0.f,0.f,0.f,0.f,0.f};
#pragma unroll
    for (int kb = 0; kb < 6; ++kb) {
      v16bf a = load_a_frag(arow, kb, kh);
      v16bf b = *(const v16bf*)(brow + kb * 32 + kh * 16);
      acc = __builtin_amdgcn_wmma_f32_16x16x32_bf16(false, a, false, b,
                                                    (short)0, acc, false, false);
    }
    // C layout: lanes 0-15 -> N=lane, M=vgpr r; lanes 16-31 -> M = 8+r.
    // Transpose through per-wave LDS (same-wave LDS is in-order, no barrier)
    // so the global write is row-major / e-contiguous -> 2x float4 per lane.
#pragma unroll
    for (int r = 0; r < 8; ++r) stg[(r + (kh ? 8 : 0)) * 16 + c4] = acc[r];
    int row = lane >> 1, half = lane & 1;
    float4 lo = *(const float4*)&stg[row * 16 + half * 8];
    float4 hi = *(const float4*)&stg[row * 16 + half * 8 + 4];
    int tok = m0 + row;
    int b_  = tok >> 12;                 // / Sz
    int s_  = tok & (Sz - 1);
    int e0  = n * 16 + half * 8;
    float* dst = &Wx[(((long)s_ * NHz + h) * 4 + g) * (Bz * DHz) + b_ * DHz + e0];
    *(float4*)dst       = lo;
    *(float4*)(dst + 4) = hi;
  }
}

// ---- Kernel 4: sequential sLSTM scan + fused GroupNorm -------------------
// One workgroup per head. R_h (288 KB bf16) is DMA'd into LDS once via the
// CDNA5 async-to-LDS path (GLOBAL_LOAD_ASYNC_TO_LDS_B128, ASYNCcnt) and then
// serves as the WMMA B operand for all 4096 serial steps. 24 waves do the
// Y(16x192)@R(192x768) WMMA; waves 0-7 (one per batch) do the pointwise
// sLSTM update with register state + fused GroupNorm.
__global__ __launch_bounds__(768) void k_scan(const float* __restrict__ Wx,
                                              const __bf16* __restrict__ rbf,
                                              const float* __restrict__ rb,
                                              const float* __restrict__ gnw,
                                              float* __restrict__ out) {
  __shared__ __bf16 Rlds[NCOL * DHz];  // 294912 B: recurrent matrix, B operand
  __shared__ __bf16 yA[16 * DHz];      // 6144 B: A operand, rows = batch (8 used)
  __shared__ float  raw[NCOL * Bz];    // 24576 B: Ry preactivations [n][b]
  const int h    = blockIdx.x;
  const int tid  = threadIdx.x;
  const int lane = tid & 31, wv = tid >> 5;
  const int c4   = lane & 15, kh = lane >> 4;

  // one-time async DMA of packed R_h into LDS: 16B per lane per issue,
  // no VGPR round-trip; completion fenced with s_wait_asynccnt then barrier.
  const __bf16* rh = rbf + (size_t)h * NCOL * DHz;
  for (int i = tid * 8; i < NCOL * DHz; i += 768 * 8) {
    unsigned ldsa = (unsigned)(uintptr_t)(lds_bf16*)&Rlds[i];
    const __bf16* ga = &rh[i];
    asm volatile("global_load_async_to_lds_b128 %0, %1, off"
                 :: "v"(ldsa), "v"(ga) : "memory");
  }
  for (int i = tid; i < 16 * DHz; i += 768) yA[i] = (__bf16)0.0f;

  float cs[6], ns[6], ms[6];
#pragma unroll
  for (int j = 0; j < 6; ++j) { cs[j] = 0.f; ns[j] = 0.f; ms[j] = 0.f; }

  float gw[6], bi[6], bf[6], bz[6], bo[6];
  if (wv < 8) {
#pragma unroll
    for (int j = 0; j < 6; ++j) {
      int e = lane + 32 * j;
      gw[j] = gnw[h * DHz + e];
      bi[j] = rb[(0 * NHz + h) * DHz + e];
      bf[j] = rb[(1 * NHz + h) * DHz + e];
      bz[j] = rb[(2 * NHz + h) * DHz + e];
      bo[j] = rb[(3 * NHz + h) * DHz + e];
    }
  }
  const int n0 = wv * 32;              // two 16-wide N tiles per wave
  asm volatile("s_wait_asynccnt 0x0" ::: "memory");
  __syncthreads();

  for (int t = 0; t < Sz; ++t) {
    // ---- phase A: Ry = Y @ R_h  (48 tiles across 24 waves) ----
    v16bf af[6];
#pragma unroll
    for (int kb = 0; kb < 6; ++kb) af[kb] = load_a_frag(&yA[c4 * DHz], kb, kh);
#pragma unroll
    for (int jt = 0; jt < 2; ++jt) {
      int nt = n0 + jt * 16;
      const __bf16* brow = &Rlds[(size_t)(nt + c4) * DHz];
      v8f acc = {0.f,0.f,0.f,0.f,0.f,0.f,0.f,0.f};
#pragma unroll
      for (int kb = 0; kb < 6; ++kb) {
        v16bf b = *(const v16bf*)(brow + kb * 32 + kh * 16);
        acc = __builtin_amdgcn_wmma_f32_16x16x32_bf16(false, af[kb], false, b,
                                                      (short)0, acc, false, false);
      }
      if (kh == 0) {                 // lanes 0-15 hold batches M=0..7
#pragma unroll
        for (int r = 0; r < 8; ++r) raw[(nt + c4) * Bz + r] = acc[r];
      }
    }
    __syncthreads();

    // ---- phase B: gates + state update + fused GroupNorm ----
    if (wv < 8) {
      const int b_ = wv;
      const float* wxb = Wx + (((long)t * NHz + h) * 4) * (Bz * DHz) + b_ * DHz;
      float yv[6], sum = 0.f, sq = 0.f;
#pragma unroll
      for (int j = 0; j < 6; ++j) {
        int e = lane + 32 * j;
        float iraw = wxb[e]                 + raw[(0 * DHz + e) * Bz + b_] + bi[j];
        float fraw = wxb[1 * (Bz*DHz) + e]  + raw[(1 * DHz + e) * Bz + b_] + bf[j];
        float zraw = wxb[2 * (Bz*DHz) + e]  + raw[(2 * DHz + e) * Bz + b_] + bz[j];
        float oraw = wxb[3 * (Bz*DHz) + e]  + raw[(3 * DHz + e) * Bz + b_] + bo[j];
        float ls   = fminf(fraw, 0.f) - log1pf(__expf(-fabsf(fraw)));  // log_sigmoid
        float lpm  = ms[j] + ls;
        float mnew = fmaxf(iraw, lpm);
        float ig   = __expf(iraw - mnew);
        float fg   = __expf(lpm - mnew);
        float cn   = fg * cs[j] + ig * tanhf(zraw);
        float nn   = fg * ns[j] + ig;
        float y    = (1.f / (1.f + __expf(-oraw))) * cn / nn;
        cs[j] = cn; ns[j] = nn; ms[j] = mnew;
        yA[b_ * DHz + e] = (__bf16)y;     // feed next step's A operand
        yv[j] = y; sum += y; sq += y * y;
      }
#pragma unroll
      for (int off = 16; off > 0; off >>= 1) {   // wave32 reduction
        sum += __shfl_xor(sum, off);
        sq  += __shfl_xor(sq,  off);
      }
      float mean = sum * (1.f / DHz);
      float inv  = rsqrtf(sq * (1.f / DHz) - mean * mean + EPSF);
      float* ob = out + ((long)b_ * Sz + t) * Dz + h * DHz;
#pragma unroll
      for (int j = 0; j < 6; ++j) {
        int e = lane + 32 * j;
        ob[e] = (yv[j] - mean) * inv * gw[j];
      }
      if (t + 1 < Sz)   // pull next step's Wx toward the caches
        __builtin_prefetch(wxb + (long)NHz * 4 * (Bz * DHz), 0, 0);
    }
    __syncthreads();
  }
}

// ---- launcher -------------------------------------------------------------
extern "C" void kernel_launch(void* const* d_in, const int* in_sizes, int n_in,
                              void* d_out, int out_size, void* d_ws, size_t ws_size,
                              hipStream_t stream) {
  (void)in_sizes; (void)n_in; (void)out_size; (void)ws_size;
  const float* x   = (const float*)d_in[0];
  const float* cw  = (const float*)d_in[1];
  const float* cb  = (const float*)d_in[2];
  const float* wf  = (const float*)d_in[3];
  const float* wi  = (const float*)d_in[4];
  const float* wz  = (const float*)d_in[5];
  const float* wo  = (const float*)d_in[6];
  const float* rk  = (const float*)d_in[7];
  const float* rb  = (const float*)d_in[8];
  const float* gnw = (const float*)d_in[9];
  float* out = (float*)d_out;

  // workspace layout
  float*  xc  = (float*)d_ws;                                   // B*S*D f32
  float*  Wx  = xc + (size_t)Bz * Sz * Dz;                      // S*NH*4*B*DH f32
  __bf16* wbf = (__bf16*)(Wx + (size_t)Sz * NHz * 4 * Bz * DHz);// 4*NH*DH*DH bf16
  __bf16* rbf = wbf + (size_t)4 * NHz * DHz * DHz;              // NH*NCOL*DH bf16

  long total = (long)Bz * Sz * Dz;
  k_conv_silu<<<(int)((total + 255) / 256), 256, 0, stream>>>(x, cw, cb, xc);
  k_pack<<<(NHz * NCOL * DHz + 255) / 256, 256, 0, stream>>>(wf, wi, wz, wo, rk, wbf, rbf);
  k_proj<<<(Bz * Sz) / 16, 256, 0, stream>>>(x, xc, wbf, Wx);
  k_scan<<<NHz, 768, 0, stream>>>(Wx, rbf, rb, gnw, out);
}
